// PCTransformer_26001732010440
// MI455X (gfx1250) — matrix operand order repaired and checked by reading, not compiled, not run on hardware
//
#include <hip/hip_runtime.h>

// ---------------------------------------------------------------------------
// PC-Transformer on MI455X (gfx1250): all GEMMs via V_WMMA_F32_16X16X32_BF16,
// fp32 accumulate, wave32. LDS tiles arranged so every WMMA fragment is two
// contiguous 16-byte chunks -> ds_load_b128; full-tile fast path removes all
// per-element bounds guards from the staging loops.
// ---------------------------------------------------------------------------

typedef __attribute__((ext_vector_type(16))) __bf16 v16bf;
typedef __attribute__((ext_vector_type(8)))  __bf16 v8bf;
typedef __attribute__((ext_vector_type(8)))  float  v8f;

constexpr int Bc = 2, Sc = 512, Dm = 512, FFm = 2048, Vm = 32000;
constexpr int Lm = 4, Hm = 8, HD = 64, TSTEPS = 5;
constexpr int BSc = Bc * Sc;                 // 1024 tokens
constexpr float LRc = 0.1f;

#define TM 64
#define TN 128
#define TK 32
#define APAD 40   // A row: 32 bf16 + pad, 80B (16B-aligned rows)
#define BPAD 40   // B^T row: 32 bf16 + pad, 80B

union Pk { unsigned u; __bf16 h[2]; };

// ---------------------------------------------------------------------------
// General strided + batched WMMA GEMM.
//   C[z; m, n] = beta*C + alpha * sum_k A[z; m,k] * B[z; k,n]
//                + biasS*bias[n] + subS*SUB[z; m,n]   (+ optional clip +-100)
// A: offA(z) + m*sAm + k*sAk      offX(z) = (z/zH)*sXb1 + (z%zH)*sXb0
// B: offB(z) + k*sBk + n*sBn ;  C/SUB: offC(z) + m*ldc + n
// Block: 256 threads = 8 wave32 waves; block tile 64x128, wave tile 32x32.
// ---------------------------------------------------------------------------
__global__ __launch_bounds__(256) void k_gemm_wmma(
    const float* __restrict__ A, int sAm, int sAk, long sAb1, long sAb0,
    const float* __restrict__ B, int sBk, int sBn, long sBb1, long sBb0,
    float* __restrict__ C, int ldc, long sCb1, long sCb0,
    const float* __restrict__ bias, const float* __restrict__ SUB,
    float alpha, float beta, float biasS, float subS,
    int M, int N, int K, int zH, int clipf)
{
    __shared__ __align__(16) __bf16 As[TM][APAD];   // row-major (m, k)
    __shared__ __align__(16) __bf16 Bs[TN][BPAD];   // TRANSPOSED (n, k)

    const int z = blockIdx.z;
    const long zq = (long)(z / zH), zr = (long)(z % zH);
    const long offA = zq * sAb1 + zr * sAb0;
    const long offB = zq * sBb1 + zr * sBb0;
    const long offC = zq * sCb1 + zr * sCb0;

    const int tid  = threadIdx.x;
    const int wave = tid >> 5;
    const int lane = tid & 31;
    const int lm   = lane & 15;
    const int lhi  = lane >> 4;
    const int wm   = (wave >> 2) * 32;       // 0 / 32
    const int wn   = (wave & 3) * 32;        // 0 / 32 / 64 / 96
    const int blockM = blockIdx.y * TM;
    const int blockN = blockIdx.x * TN;
    const bool fullT = (blockM + TM <= M) && (blockN + TN <= N);

    v8f acc[2][2];
#pragma unroll
    for (int i = 0; i < 2; ++i)
#pragma unroll
        for (int j = 0; j < 2; ++j)
#pragma unroll
            for (int r = 0; r < 8; ++r) acc[i][j][r] = 0.0f;

    for (int k0 = 0; k0 < K; k0 += TK) {
        if (fullT && (k0 + TK <= K)) {
            // -------- fast path: no per-element guards --------
            if (sAk == 1) {
#pragma unroll
                for (int it = 0; it < (TM * TK / 2) / 256; ++it) {
                    const int p = tid + it * 256;
                    const int c2 = (p & 15) * 2, r = p >> 4;
                    const float2 v = *(const float2*)&A[offA + (long)(blockM + r) * sAm + (k0 + c2)];
                    Pk pk; pk.h[0] = (__bf16)v.x; pk.h[1] = (__bf16)v.y;
                    *(unsigned*)&As[r][c2] = pk.u;
                }
            } else {
#pragma unroll
                for (int it = 0; it < (TM * TK / 2) / 256; ++it) {
                    const int p = tid + it * 256;
                    const int c2 = (p & 15) * 2, r = p >> 4;
                    const long base = offA + (long)(blockM + r) * sAm + (long)(k0 + c2) * sAk;
                    Pk pk; pk.h[0] = (__bf16)A[base]; pk.h[1] = (__bf16)A[base + sAk];
                    *(unsigned*)&As[r][c2] = pk.u;
                }
            }
            if (sBk == 1) {
#pragma unroll
                for (int it = 0; it < (TN * TK / 2) / 256; ++it) {
                    const int p = tid + it * 256;
                    const int n = p & 127, k2 = (p >> 7) * 2;
                    const float2 v = *(const float2*)&B[offB + (long)(blockN + n) * sBn + (k0 + k2)];
                    Pk pk; pk.h[0] = (__bf16)v.x; pk.h[1] = (__bf16)v.y;
                    *(unsigned*)&Bs[n][k2] = pk.u;
                }
            } else {
#pragma unroll
                for (int it = 0; it < (TN * TK / 2) / 256; ++it) {
                    const int p = tid + it * 256;
                    const int n = p & 127, k2 = (p >> 7) * 2;
                    const long base = offB + (long)(blockN + n) * sBn + (long)(k0 + k2) * sBk;
                    Pk pk; pk.h[0] = (__bf16)B[base]; pk.h[1] = (__bf16)B[base + sBk];
                    *(unsigned*)&Bs[n][k2] = pk.u;
                }
            }
        } else {
            // -------- guarded path (edge tiles, e.g. N=64 attention) --------
            for (int p = tid; p < TM * (TK / 2); p += 256) {
                const int c2 = (p & 15) * 2, r = p >> 4;
                const int gm = blockM + r, gk = k0 + c2;
                float v0 = 0.0f, v1 = 0.0f;
                if (gm < M) {
                    const long base = offA + (long)gm * sAm;
                    if (gk < K)     v0 = A[base + (long)gk * sAk];
                    if (gk + 1 < K) v1 = A[base + (long)(gk + 1) * sAk];
                }
                Pk pk; pk.h[0] = (__bf16)v0; pk.h[1] = (__bf16)v1;
                *(unsigned*)&As[r][c2] = pk.u;
            }
            for (int p = tid; p < TN * (TK / 2); p += 256) {
                const int n = p & 127, k2 = (p >> 7) * 2;
                const int gn = blockN + n, gk = k0 + k2;
                float v0 = 0.0f, v1 = 0.0f;
                if (gn < N) {
                    const long base = offB + (long)gn * sBn;
                    if (gk < K)     v0 = B[base + (long)gk * sBk];
                    if (gk + 1 < K) v1 = B[base + (long)(gk + 1) * sBk];
                }
                Pk pk; pk.h[0] = (__bf16)v0; pk.h[1] = (__bf16)v1;
                *(unsigned*)&Bs[n][k2] = pk.u;
            }
        }
        __syncthreads();

        // Fragments: two ds_load_b128 each (contiguous 16B chunks).
        // A lane(L): m=lm, K = {kb..kb+7} ++ {16+kb..16+kb+7}, kb = 8*lhi.
        v16bf afrag[2], bfrag[2];
#pragma unroll
        for (int t = 0; t < 2; ++t) {
            const int mrow = wm + t * 16 + lm;
            const int kb = lhi * 8;
            v8bf lo = *(const v8bf*)&As[mrow][kb];
            v8bf hi = *(const v8bf*)&As[mrow][16 + kb];
            afrag[t] = __builtin_shufflevector(lo, hi, 0, 1, 2, 3, 4, 5, 6, 7,
                                               8, 9, 10, 11, 12, 13, 14, 15);
        }
        // B lane(L): n=lm, K = {kb..kb+7} ++ {16+kb..16+kb+7}, kb = 8*lhi.
#pragma unroll
        for (int t = 0; t < 2; ++t) {
            const int ncol = wn + t * 16 + lm;
            const int kb = lhi * 8;
            v8bf lo = *(const v8bf*)&Bs[ncol][kb];
            v8bf hi = *(const v8bf*)&Bs[ncol][16 + kb];
            bfrag[t] = __builtin_shufflevector(lo, hi, 0, 1, 2, 3, 4, 5, 6, 7,
                                               8, 9, 10, 11, 12, 13, 14, 15);
        }
#pragma unroll
        for (int i = 0; i < 2; ++i)
#pragma unroll
            for (int j = 0; j < 2; ++j)
                acc[i][j] = __builtin_amdgcn_wmma_f32_16x16x32_bf16(
                    false, afrag[i], false, bfrag[j], (short)0, acc[i][j],
                    false, false);
        __syncthreads();
    }

    // Epilogue. C/D layout: VGPR r -> M = r + 8*(lane>=16), N = lane&15.
#pragma unroll
    for (int i = 0; i < 2; ++i)
#pragma unroll
        for (int j = 0; j < 2; ++j)
#pragma unroll
            for (int r = 0; r < 8; ++r) {
                const int m = blockM + wm + i * 16 + r + 8 * lhi;
                const int n = blockN + wn + j * 16 + lm;
                if (m < M && n < N) {
                    const long idx = offC + (long)m * ldc + n;
                    float v = alpha * acc[i][j][r];
                    if (bias) v += biasS * bias[n];
                    if (SUB)  v += subS * SUB[idx];
                    if (beta != 0.0f) v += beta * C[idx];
                    if (clipf) v = fminf(fmaxf(v, -100.0f), 100.0f);
                    C[idx] = v;
                }
            }
}

// ---------------------------------------------------------------------------
// Elementwise / reduction kernels
// ---------------------------------------------------------------------------
__device__ __forceinline__ float gelu_f(float x) {
    float u = 0.7978845608028654f * (x + 0.044715f * x * x * x);
    return 0.5f * x * (1.0f + tanhf(u));
}
__device__ __forceinline__ float gelu_g(float x) {
    float x2 = x * x;
    float u = 0.7978845608028654f * (x + 0.044715f * x2 * x);
    float t = tanhf(u);
    float du = 0.7978845608028654f * (1.0f + 0.134145f * x2);
    return 0.5f * (1.0f + t) + 0.5f * x * (1.0f - t * t) * du;
}

__global__ void k_zero(float* p, long n) {
    long i = (long)blockIdx.x * blockDim.x + threadIdx.x;
    if (i < n) p[i] = 0.0f;
}
__global__ void k_copy(float* d, const float* s, long n) {
    long i = (long)blockIdx.x * blockDim.x + threadIdx.x;
    if (i < n) d[i] = s[i];
}
__global__ void k_emb(const int* __restrict__ ids, const float* __restrict__ we,
                      const float* __restrict__ pe, float* __restrict__ out) {
    long i = (long)blockIdx.x * blockDim.x + threadIdx.x;
    if (i < (long)BSc * Dm) {
        int d = (int)(i % Dm);
        long t = i / Dm;
        int s = (int)(t % Sc);
        out[i] = we[(long)ids[t] * Dm + d] + pe[(long)s * Dm + d];
    }
}
__global__ void k_gelu(float* out, const float* in, long n) {
    long i = (long)blockIdx.x * blockDim.x + threadIdx.x;
    if (i < n) out[i] = gelu_f(in[i]);
}
__global__ void k_gelu_sub(float* e1, const float* x2, const float* pre, long n) {
    long i = (long)blockIdx.x * blockDim.x + threadIdx.x;
    if (i < n) e1[i] = x2[i] - gelu_f(pre[i]);
}
__global__ void k_gelu_vjp(float* out, const float* e1, const float* pre, long n) {
    long i = (long)blockIdx.x * blockDim.x + threadIdx.x;
    if (i < n) out[i] = e1[i] * gelu_g(pre[i]);
}
__global__ void k_pc_update(float* x, const float* g, const float* s, long n) {
    long i = (long)blockIdx.x * blockDim.x + threadIdx.x;
    if (i < n) x[i] += LRc * (g[i] - s[i]);
}
__global__ void k_sub(float* out, const float* a, const float* b, long n) {
    long i = (long)blockIdx.x * blockDim.x + threadIdx.x;
    if (i < n) out[i] = a[i] - b[i];
}
__global__ void k_scatter_tgt(float* eo, const int* t, int n) {
    int i = blockIdx.x * blockDim.x + threadIdx.x;
    if (i < n) eo[(long)i * Vm + t[i]] += 1.0f;
}

// out[row,:] = LN(x[row,:])*g + b - sub[row,:]
__global__ __launch_bounds__(256) void k_ln_sub(
    const float* __restrict__ x, const float* __restrict__ g,
    const float* __restrict__ b, const float* __restrict__ sub,
    float* __restrict__ out, int D)
{
    __shared__ float red[256];
    const int tid = threadIdx.x;
    const long row = blockIdx.x;
    const float* xr = x + row * D;
    float s = 0.0f;
    for (int i = tid; i < D; i += 256) s += xr[i];
    red[tid] = s; __syncthreads();
    for (int st = 128; st > 0; st >>= 1) { if (tid < st) red[tid] += red[tid + st]; __syncthreads(); }
    const float mean = red[0] / D;
    __syncthreads();
    float v = 0.0f;
    for (int i = tid; i < D; i += 256) { float d = xr[i] - mean; v += d * d; }
    red[tid] = v; __syncthreads();
    for (int st = 128; st > 0; st >>= 1) { if (tid < st) red[tid] += red[tid + st]; __syncthreads(); }
    const float rstd = rsqrtf(red[0] / D + 1e-5f);
    for (int i = tid; i < D; i += 256)
        out[row * D + i] = (xr[i] - mean) * rstd * g[i] + b[i] - sub[row * D + i];
}

// Causal softmax over rows of (B,H,S,S); masked entries -> 0
__global__ __launch_bounds__(256) void k_softmax_causal(float* a, int S)
{
    __shared__ float red[256];
    const int tid = threadIdx.x;
    const long row = blockIdx.x;
    const int q = (int)(row % S);
    float* ar = a + row * (long)S;
    const int valid = q + 1;
    float mx = -1e30f;
    for (int i = tid; i < valid; i += 256) mx = fmaxf(mx, ar[i]);
    red[tid] = mx; __syncthreads();
    for (int st = 128; st > 0; st >>= 1) { if (tid < st) red[tid] = fmaxf(red[tid], red[tid + st]); __syncthreads(); }
    mx = red[0];
    __syncthreads();
    float sm = 0.0f;
    for (int i = tid; i < valid; i += 256) sm += expf(ar[i] - mx);
    red[tid] = sm; __syncthreads();
    for (int st = 128; st > 0; st >>= 1) { if (tid < st) red[tid] += red[tid + st]; __syncthreads(); }
    const float inv = 1.0f / red[0];
    for (int i = tid; i < S; i += 256)
        ar[i] = (i < valid) ? expf(ar[i] - mx) * inv : 0.0f;
}

// dsc = a * (da - sum_k(da*a)) * scale   (in place in da)
__global__ __launch_bounds__(256) void k_softmax_bwd(
    float* __restrict__ da, const float* __restrict__ a, int S, float scale)
{
    __shared__ float red[256];
    const int tid = threadIdx.x;
    const long row = blockIdx.x;
    float* dr = da + row * (long)S;
    const float* ar = a + row * (long)S;
    float s = 0.0f;
    for (int i = tid; i < S; i += 256) s += dr[i] * ar[i];
    red[tid] = s; __syncthreads();
    for (int st = 128; st > 0; st >>= 1) { if (tid < st) red[tid] += red[tid + st]; __syncthreads(); }
    const float tot = red[0];
    for (int i = tid; i < S; i += 256) dr[i] = ar[i] * (dr[i] - tot) * scale;
}

// ---------------------------------------------------------------------------
// Host-side helpers
// ---------------------------------------------------------------------------
static void gemm(hipStream_t st,
                 const float* A, int sAm, int sAk, long sAb1, long sAb0,
                 const float* B, int sBk, int sBn, long sBb1, long sBb0,
                 float* C, int ldc, long sCb1, long sCb0,
                 const float* bias, const float* SUB,
                 float alpha, float beta, float biasS, float subS,
                 int M, int N, int K, int nz, int zH, int clipf)
{
    dim3 grid((N + TN - 1) / TN, (M + TM - 1) / TM, nz);
    k_gemm_wmma<<<grid, 256, 0, st>>>(A, sAm, sAk, sAb1, sAb0,
                                      B, sBk, sBn, sBb1, sBb0,
                                      C, ldc, sCb1, sCb0, bias, SUB,
                                      alpha, beta, biasS, subS,
                                      M, N, K, zH, clipf);
}

#define EWG(n) dim3((unsigned)(((n) + 255) / 256)), dim3(256), 0, stream

extern "C" void kernel_launch(void* const* d_in, const int* in_sizes, int n_in,
                              void* d_out, int out_size, void* d_ws, size_t ws_size,
                              hipStream_t stream)
{
    (void)in_sizes; (void)n_in; (void)out_size; (void)ws_size;

    const int*   target_ids = (const int*)d_in[0];
    const int*   input_ids  = (const int*)d_in[1];
    const float* word_emb   = (const float*)d_in[2];
    const float* pos_emb    = (const float*)d_in[3];
    const float* Wq = (const float*)d_in[4];  const float* bq = (const float*)d_in[5];
    const float* Wk = (const float*)d_in[6];  const float* bk = (const float*)d_in[7];
    const float* Wv = (const float*)d_in[8];  const float* bv = (const float*)d_in[9];
    const float* Wo = (const float*)d_in[10]; const float* bo = (const float*)d_in[11];
    const float* ln1_g = (const float*)d_in[12]; const float* ln1_b = (const float*)d_in[13];
    const float* W1 = (const float*)d_in[14]; const float* b1 = (const float*)d_in[15];
    const float* W2 = (const float*)d_in[16]; const float* b2 = (const float*)d_in[17];
    const float* ln2_g = (const float*)d_in[18]; const float* ln2_b = (const float*)d_in[19];
    const float* Wout = (const float*)d_in[20]; const float* bout = (const float*)d_in[21];
    float* out = (float*)d_out;

    const long nD  = (long)BSc * Dm;
    const long nFF = (long)BSc * FFm;
    const long nV  = (long)BSc * Vm;
    const long SD  = (long)Sc * Dm;
    const long SS  = (long)Sc * Sc;
    const long HSS = (long)Hm * SS;
    const long nA  = (long)Bc * HSS;

    float* ws = (float*)d_ws;
    size_t wo = 0;
    auto alloc = [&](long n) { float* p = ws + wo; wo += (size_t)n; return p; };

    float* emb = alloc(nD);
    float* xq  = alloc((long)Lm * nD);
    float* xa  = alloc((long)Lm * nD);
    float* x1  = alloc((long)Lm * nD);
    float* x2  = alloc((long)Lm * nFF);
    float* xo  = alloc(nD);
    struct EB { float *eq, *ea, *e1, *e2, *eemb; };
    EB eb[2];
    for (int i = 0; i < 2; ++i) {
        eb[i].eq = alloc((long)Lm * nD);  eb[i].ea = alloc((long)Lm * nD);
        eb[i].e1 = alloc((long)Lm * nFF); eb[i].e2 = alloc((long)Lm * nD);
        eb[i].eemb = alloc(nD);
    }
    float* e_out = alloc(nV);
    float* qb = alloc((long)Lm * nD);
    float* kb = alloc((long)Lm * nD);
    float* vb = alloc((long)Lm * nD);
    float* ab = alloc((long)Lm * nA);
    float* pre1s = alloc((long)Lm * nFF);
    float* tmpD  = alloc(nD);
    float* tmpFF = alloc(nFF);
    float* dsc = alloc(nA);
    float* dq = alloc(nD); float* dk = alloc(nD); float* dv = alloc(nD);
    float* gq = alloc(nD);

    // ep(t=0) = zeros (eb[1] region is contiguous by construction)
    const long EPSZ = 3 * (long)Lm * nD + (long)Lm * nFF + nD;
    k_zero<<<EWG(EPSZ)>>>(eb[1].eq, EPSZ);

    // MHA forward: fills q/k/v/a buffers for layer l; output either plain
    // (subX==nullptr) or eq = subX - attn_out fused in the GEMM epilogue.
    auto mha_fwd = [&](const float* xin, int l, float* outC, const float* subX) {
        const float* Wq_l = Wq + (long)l * Dm * Dm; const float* bq_l = bq + (long)l * Dm;
        const float* Wk_l = Wk + (long)l * Dm * Dm; const float* bk_l = bk + (long)l * Dm;
        const float* Wv_l = Wv + (long)l * Dm * Dm; const float* bv_l = bv + (long)l * Dm;
        float* qd = qb + (long)l * nD; float* kd = kb + (long)l * nD;
        float* vd = vb + (long)l * nD; float* ad = ab + (long)l * nA;
        gemm(stream, xin, Dm, 1, 0, 0, Wq_l, 1, Dm, 0, 0, qd, Dm, 0, 0,
             bq_l, nullptr, 1.f, 0.f, 1.f, 0.f, BSc, Dm, Dm, 1, 1, 0);
        gemm(stream, xin, Dm, 1, 0, 0, Wk_l, 1, Dm, 0, 0, kd, Dm, 0, 0,
             bk_l, nullptr, 1.f, 0.f, 1.f, 0.f, BSc, Dm, Dm, 1, 1, 0);
        gemm(stream, xin, Dm, 1, 0, 0, Wv_l, 1, Dm, 0, 0, vd, Dm, 0, 0,
             bv_l, nullptr, 1.f, 0.f, 1.f, 0.f, BSc, Dm, Dm, 1, 1, 0);
        // scores: per (b,h) GEMM, scale 1/sqrt(hd) = 1/8
        gemm(stream, qd, Dm, 1, SD, HD, kd, 1, Dm, SD, HD, ad, Sc, HSS, SS,
             nullptr, nullptr, 0.125f, 0.f, 0.f, 0.f, Sc, Sc, HD, Bc * Hm, Hm, 0);
        k_softmax_causal<<<dim3(Bc * Hm * Sc), dim3(256), 0, stream>>>(ad, Sc);
        if (subX)
            gemm(stream, ad, Sc, 1, HSS, SS, vd, Dm, 1, SD, HD, outC, Dm, SD, HD,
                 nullptr, subX, -1.f, 0.f, 0.f, 1.f, Sc, HD, Sc, Bc * Hm, Hm, 0);
        else
            gemm(stream, ad, Sc, 1, HSS, SS, vd, Dm, 1, SD, HD, outC, Dm, SD, HD,
                 nullptr, nullptr, 1.f, 0.f, 0.f, 0.f, Sc, HD, Sc, Bc * Hm, Hm, 0);
    };

    // ---- embedding + feed-forward init sweep (xs0) ----
    k_emb<<<EWG(nD)>>>(input_ids, word_emb, pos_emb, emb);
    k_copy<<<EWG(nD)>>>(xq, emb, nD);
    for (int l = 0; l < Lm; ++l) {
        const float* Wo_l = Wo + (long)l * Dm * Dm; const float* bo_l = bo + (long)l * Dm;
        const float* W1_l = W1 + (long)l * FFm * Dm; const float* b1_l = b1 + (long)l * FFm;
        const float* W2_l = W2 + (long)l * Dm * FFm; const float* b2_l = b2 + (long)l * Dm;
        mha_fwd(xq + (long)l * nD, l, xa + (long)l * nD, nullptr);
        gemm(stream, xa + (long)l * nD, Dm, 1, 0, 0, Wo_l, 1, Dm, 0, 0,
             x1 + (long)l * nD, Dm, 0, 0, bo_l, nullptr, 1.f, 0.f, 1.f, 0.f,
             BSc, Dm, Dm, 1, 1, 0);
        gemm(stream, x1 + (long)l * nD, Dm, 1, 0, 0, W1_l, 1, Dm, 0, 0,
             tmpFF, FFm, 0, 0, b1_l, nullptr, 1.f, 0.f, 1.f, 0.f,
             BSc, FFm, Dm, 1, 1, 0);
        k_gelu<<<EWG(nFF)>>>(x2 + (long)l * nFF, tmpFF, nFF);
        float* nxt = (l < Lm - 1) ? xq + (long)(l + 1) * nD : xo;
        gemm(stream, x2 + (long)l * nFF, FFm, 1, 0, 0, W2_l, 1, FFm, 0, 0,
             nxt, Dm, 0, 0, b2_l, nullptr, 1.f, 0.f, 1.f, 0.f,
             BSc, Dm, FFm, 1, 1, 0);
    }

    // ---- T Jacobi PC steps ----
    for (int t = 0; t < TSTEPS; ++t) {
        EB& E = eb[t & 1];
        EB& P = eb[1 - (t & 1)];

        // errors(xs)
        gemm(stream, xo, Dm, 1, 0, 0, Wout, 1, Dm, 0, 0, e_out, Vm, 0, 0,
             bout, nullptr, -1.f, 0.f, -1.f, 0.f, BSc, Vm, Dm, 1, 1, 0);
        k_scatter_tgt<<<EWG(BSc)>>>(e_out, target_ids, BSc);
        for (int l = 0; l < Lm; ++l) {
            const float* Wo_l = Wo + (long)l * Dm * Dm; const float* bo_l = bo + (long)l * Dm;
            const float* W1_l = W1 + (long)l * FFm * Dm; const float* b1_l = b1 + (long)l * FFm;
            const float* W2_l = W2 + (long)l * Dm * FFm; const float* b2_l = b2 + (long)l * Dm;
            const float* nt = (l < Lm - 1) ? xq + (long)(l + 1) * nD : xo;
            // e2 = LN(nt) - (x2@W2^T + b2)
            gemm(stream, x2 + (long)l * nFF, FFm, 1, 0, 0, W2_l, 1, FFm, 0, 0,
                 tmpD, Dm, 0, 0, b2_l, nullptr, 1.f, 0.f, 1.f, 0.f, BSc, Dm, FFm, 1, 1, 0);
            k_ln_sub<<<dim3(BSc), dim3(256), 0, stream>>>(nt, ln2_g + (long)l * Dm,
                ln2_b + (long)l * Dm, tmpD, E.e2 + (long)l * nD, Dm);
            // pre1 (saved) and e1 = x2 - gelu(pre1)
            gemm(stream, x1 + (long)l * nD, Dm, 1, 0, 0, W1_l, 1, Dm, 0, 0,
                 pre1s + (long)l * nFF, FFm, 0, 0, b1_l, nullptr, 1.f, 0.f, 1.f, 0.f,
                 BSc, FFm, Dm, 1, 1, 0);
            k_gelu_sub<<<EWG(nFF)>>>(E.e1 + (long)l * nFF, x2 + (long)l * nFF,
                                     pre1s + (long)l * nFF, nFF);
            // ea = LN(x1) - (xa@Wo^T + bo)
            gemm(stream, xa + (long)l * nD, Dm, 1, 0, 0, Wo_l, 1, Dm, 0, 0,
                 tmpD, Dm, 0, 0, bo_l, nullptr, 1.f, 0.f, 1.f, 0.f, BSc, Dm, Dm, 1, 1, 0);
            k_ln_sub<<<dim3(BSc), dim3(256), 0, stream>>>(x1 + (long)l * nD,
                ln1_g + (long)l * Dm, ln1_b + (long)l * Dm, tmpD,
                E.ea + (long)l * nD, Dm);
            // eq = xa - mha(xq)  (fused in apply GEMM epilogue); saves q,k,v,a
            mha_fwd(xq + (long)l * nD, l, E.eq + (long)l * nD, xa + (long)l * nD);
        }
        k_sub<<<EWG(nD)>>>(E.eemb, xq, emb, nD);

        // update(xs, e, ep) in place (Jacobi: all RHS terms already stored)
        gemm(stream, e_out, Vm, 1, 0, 0, Wout, Dm, 1, 0, 0, xo, Dm, 0, 0,
             nullptr, P.e2 + (long)(Lm - 1) * nD, LRc, 1.f, 0.f, -LRc,
             BSc, Dm, Vm, 1, 1, 0);
        for (int l = 0; l < Lm; ++l) {
            const float* Wq_l = Wq + (long)l * Dm * Dm;
            const float* Wk_l = Wk + (long)l * Dm * Dm;
            const float* Wv_l = Wv + (long)l * Dm * Dm;
            const float* Wo_l = Wo + (long)l * Dm * Dm;
            const float* W1_l = W1 + (long)l * FFm * Dm;
            const float* W2_l = W2 + (long)l * Dm * FFm;
            // x2 += LR*(e2@W2 - ep1)
            gemm(stream, E.e2 + (long)l * nD, Dm, 1, 0, 0, W2_l, FFm, 1, 0, 0,
                 x2 + (long)l * nFF, FFm, 0, 0, nullptr, P.e1 + (long)l * nFF,
                 LRc, 1.f, 0.f, -LRc, BSc, FFm, Dm, 1, 1, 0);
            // x1 += LR*((e1 * gelu'(pre1)) @ W1 - epa)
            k_gelu_vjp<<<EWG(nFF)>>>(tmpFF, E.e1 + (long)l * nFF,
                                     pre1s + (long)l * nFF, nFF);
            gemm(stream, tmpFF, FFm, 1, 0, 0, W1_l, Dm, 1, 0, 0,
                 x1 + (long)l * nD, Dm, 0, 0, nullptr, P.ea + (long)l * nD,
                 LRc, 1.f, 0.f, -LRc, BSc, Dm, FFm, 1, 1, 0);
            // xa += LR*(ea@Wo - epq)
            gemm(stream, E.ea + (long)l * nD, Dm, 1, 0, 0, Wo_l, Dm, 1, 0, 0,
                 xa + (long)l * nD, Dm, 0, 0, nullptr, P.eq + (long)l * nD,
                 LRc, 1.f, 0.f, -LRc, BSc, Dm, Dm, 1, 1, 0);
            // MHA VJP at xq[l] with cotangent eq[l] (uses saved q,k,v,a)
            float* qd = qb + (long)l * nD; float* kd = kb + (long)l * nD;
            float* vd = vb + (long)l * nD; float* ad = ab + (long)l * nA;
            const float* eqc = E.eq + (long)l * nD;
            // da = eq . v^T  per (b,h)
            gemm(stream, eqc, Dm, 1, SD, HD, vd, 1, Dm, SD, HD, dsc, Sc, HSS, SS,
                 nullptr, nullptr, 1.f, 0.f, 0.f, 0.f, Sc, Sc, HD, Bc * Hm, Hm, 0);
            k_softmax_bwd<<<dim3(Bc * Hm * Sc), dim3(256), 0, stream>>>(dsc, ad, Sc, 0.125f);
            // dq = dsc @ k ; dk = dsc^T @ q ; dv = a^T @ eq
            gemm(stream, dsc, Sc, 1, HSS, SS, kd, Dm, 1, SD, HD, dq, Dm, SD, HD,
                 nullptr, nullptr, 1.f, 0.f, 0.f, 0.f, Sc, HD, Sc, Bc * Hm, Hm, 0);
            gemm(stream, dsc, 1, Sc, HSS, SS, qd, Dm, 1, SD, HD, dk, Dm, SD, HD,
                 nullptr, nullptr, 1.f, 0.f, 0.f, 0.f, Sc, HD, Sc, Bc * Hm, Hm, 0);
            gemm(stream, ad, 1, Sc, HSS, SS, eqc, Dm, 1, SD, HD, dv, Dm, SD, HD,
                 nullptr, nullptr, 1.f, 0.f, 0.f, 0.f, Sc, HD, Sc, Bc * Hm, Hm, 0);
            // gq = dq@Wq + dk@Wk + dv@Wv
            gemm(stream, dq, Dm, 1, 0, 0, Wq_l, Dm, 1, 0, 0, gq, Dm, 0, 0,
                 nullptr, nullptr, 1.f, 0.f, 0.f, 0.f, BSc, Dm, Dm, 1, 1, 0);
            gemm(stream, dk, Dm, 1, 0, 0, Wk_l, Dm, 1, 0, 0, gq, Dm, 0, 0,
                 nullptr, nullptr, 1.f, 1.f, 0.f, 0.f, BSc, Dm, Dm, 1, 1, 0);
            gemm(stream, dv, Dm, 1, 0, 0, Wv_l, Dm, 1, 0, 0, gq, Dm, 0, 0,
                 nullptr, nullptr, 1.f, 1.f, 0.f, 0.f, BSc, Dm, Dm, 1, 1, 0);
            const float* td = (l == 0) ? P.eemb : P.e2 + (long)(l - 1) * nD;
            k_pc_update<<<EWG(nD)>>>(xq + (long)l * nD, gq, td, nD);
        }
    }

    // logits = clip(xo @ Wout^T + bout, -100, 100)
    gemm(stream, xo, Dm, 1, 0, 0, Wout, 1, Dm, 0, 0, out, Vm, 0, 0,
         bout, nullptr, 1.f, 0.f, 1.f, 0.f, BSc, Vm, Dm, 1, 1, 1);
}
